// Lis_autoencoder_188978561286
// MI455X (gfx1250) — matrix operand order repaired and checked by reading, not compiled, not run on hardware
//
#include <hip/hip_runtime.h>

#define NN 8192
#define WPR 256          // 8192 bits / 32 = words per adjacency row
#define FSCALE 64.0f     // f16 dynamic-range lift
#define FINV   (1.0f / 64.0f)

typedef __attribute__((ext_vector_type(16))) _Float16 v16h;
typedef __attribute__((ext_vector_type(8)))  _Float16 v8h;
typedef __attribute__((ext_vector_type(8)))  float    v8f;

__device__ __forceinline__ float lrelu(float v) { return v > 0.0f ? v : 0.01f * v; }

// ---------------------------------------------------------------------------
// Pass 1: pack A = (edge != 0) into bit-rows, compute dinv = rsqrt(deg+1).
// ---------------------------------------------------------------------------
__global__ void pack_deg_kernel(const int* __restrict__ edge,
                                unsigned* __restrict__ aw,
                                float* __restrict__ dinv) {
  __shared__ int red[256];
  const int row = blockIdx.x;
  const int t = threadIdx.x;
  const int4* p4 = reinterpret_cast<const int4*>(edge + (size_t)row * NN) + t * 8;
  unsigned w = 0;
  int cnt = 0;
#pragma unroll
  for (int q = 0; q < 8; ++q) {
    int4 v = p4[q];
    unsigned b0 = (v.x != 0), b1 = (v.y != 0), b2 = (v.z != 0), b3 = (v.w != 0);
    w |= (b0 << (q * 4)) | (b1 << (q * 4 + 1)) | (b2 << (q * 4 + 2)) | (b3 << (q * 4 + 3));
    cnt += (int)(b0 + b1 + b2 + b3);
  }
  aw[(size_t)row * WPR + t] = w;
  red[t] = cnt;
  __syncthreads();
  for (int s = 128; s > 0; s >>= 1) {
    if (t < s) red[t] += red[t + s];
    __syncthreads();
  }
  if (t == 0) dinv[row] = rsqrtf((float)red[0] + 1.0f);   // +1 for self loop
}

// ---------------------------------------------------------------------------
// Bt[c][j] = f16( FSCALE * dinv[j] * (X @ W)[j][c] )   (transposed for WMMA B)
// ---------------------------------------------------------------------------
template <int CIN, int K>
__global__ void xw_bt_kernel(const float* __restrict__ X,
                             const float* __restrict__ W,
                             const float* __restrict__ dinv,
                             _Float16* __restrict__ Bt) {
  const int idx = blockIdx.x * blockDim.x + threadIdx.x;   // N*K threads
  const int j = idx / K;
  const int c = idx % K;
  float acc = 0.0f;
#pragma unroll 8
  for (int i = 0; i < CIN; ++i) acc = fmaf(X[j * CIN + i], W[i * K + c], acc);
  Bt[(size_t)c * NN + j] = (_Float16)(FSCALE * dinv[j] * acc);
}

// ---------------------------------------------------------------------------
// out[i][c] = lrelu( dinv[i] * sum_j (A+I)_ij * Bt[c][j] / FSCALE + bias[c] )
// Block = 8 waves over one 16-row strip; each wave covers k-range NN/8 with
// NT = K/16 column tiles sharing one decoded A fragment (NT WMMAs / decode).
// 8 partial accumulators reduced through padded LDS; epilogue split 8 ways.
// ---------------------------------------------------------------------------
template <int K>
__global__ void spmm_kernel(const unsigned* __restrict__ aw,
                            const _Float16* __restrict__ Bt,
                            const float* __restrict__ dinv,
                            const float* __restrict__ bias,
                            float* __restrict__ out,
                            _Float16* __restrict__ out16) {
  constexpr int NT = K / 16;              // 4 (K=64) or 8 (K=128) column tiles
  constexpr int KS = 8;                   // k-split waves per block
  constexpr int LRS = NT * 8 + 4;         // padded per-lane stride (floats)
  __shared__ float lred[KS * 32 * LRS];
  const int lane = threadIdx.x & 31;
  const int wv = threadIdx.x >> 5;        // 0..7, splits k-range
  const int i0 = blockIdx.x * 16;
  const int mrow = lane & 15;
  const int hi = lane >> 4;               // lane half
  const int base0 = hi * 16;              // A-frag K offsets (contiguous k = 16h + s)
  const int base1 = base0 + 8;
  const int row = i0 + mrow;
  const unsigned* awrow = aw + (size_t)row * WPR;
  const int diag_kb = i0 & ~31;           // all 16 rows share one k-slab
  const int d = row & 31;

  v8f zero = {};
  v8f acc[NT];
#pragma unroll
  for (int t = 0; t < NT; ++t) acc[t] = zero;

  const int kbeg = wv * (NN / KS);
  const int kend = kbeg + (NN / KS);
  for (int kb = kbeg; kb < kend; kb += 32) {
    const unsigned w = awrow[kb >> 5];
    v16h a;
#pragma unroll
    for (int h = 0; h < 8; ++h) {
      a[h]     = (_Float16)(float)((w >> (base0 + h)) & 1u);
      a[h + 8] = (_Float16)(float)((w >> (base1 + h)) & 1u);
    }
    if (kb == diag_kb) {                  // uniform (scalar) branch, +I
#pragma unroll
      for (int h = 0; h < 8; ++h) {
        if (base0 + h == d) a[h]     = a[h]     + (_Float16)1.0f;
        if (base1 + h == d) a[h + 8] = a[h + 8] + (_Float16)1.0f;
      }
    }
#pragma unroll
    for (int t = 0; t < NT; ++t) {
      const v16h b = *(const v16h*)(Bt + (size_t)(t * 16 + mrow) * NN + kb + hi * 16);
      acc[t] = __builtin_amdgcn_wmma_f32_16x16x32_f16(false, a, false, b,
                                                      (short)0, acc[t], false, false);
    }
  }

  // k-split reduction through LDS (padded stride vs 64-bank conflicts)
  float* mine = lred + (size_t)(wv * 32 + lane) * LRS;
#pragma unroll
  for (int t = 0; t < NT; ++t)
#pragma unroll
    for (int r = 0; r < 8; ++r) mine[t * 8 + r] = acc[t][r];
  __syncthreads();

  // NT (t,r)-units per wave: unit = wv*NT + q  ->  t = unit>>3, r = unit&7
#pragma unroll
  for (int q = 0; q < NT; ++q) {
    const int unit = wv * NT + q;
    const int t = unit >> 3;
    const int r = unit & 7;
    float s = 0.0f;
#pragma unroll
    for (int w2 = 0; w2 < KS; ++w2)
      s += lred[(size_t)(w2 * 32 + lane) * LRS + t * 8 + r];
    const int orow = i0 + r + hi * 8;     // C/D layout: M = r (+8 hi lanes)
    const int ocol = t * 16 + mrow;       // N = lane & 15
    const float v = lrelu(dinv[orow] * s * FINV + bias[ocol]);
    out[(size_t)orow * K + ocol] = v;
    if (out16) out16[(size_t)orow * K + ocol] = (_Float16)(v * FSCALE);
  }
}

// ---------------------------------------------------------------------------
// recon[i][j] = sigmoid( (re @ re^T)[i][j] ), re16 scaled by FSCALE (K = 64).
// Each wave: one 16-row A fragment shared across 4 j-tiles (8 WMMAs / wave).
// ---------------------------------------------------------------------------
__global__ void recon_kernel(const _Float16* __restrict__ re16,
                             float* __restrict__ out) {
  const int lane = threadIdx.x & 31;
  const int wv = threadIdx.x >> 5;
  const int i0 = blockIdx.y * 16;
  const int j0 = blockIdx.x * 256 + wv * 64;   // 4 j-tiles of 16
  const int mrow = lane & 15;
  const int hi = lane >> 4;
  const int base0 = hi * 16;
  const _Float16* arow = re16 + (size_t)(i0 + mrow) * 64;

  v8f zero = {};
  v8f acc[4];
#pragma unroll
  for (int t = 0; t < 4; ++t) acc[t] = zero;

#pragma unroll
  for (int kb = 0; kb < 64; kb += 32) {
    const v8h lo = *(const v8h*)(arow + kb + base0);
    const v8h hh = *(const v8h*)(arow + kb + base0 + 8);
    v16h a;
#pragma unroll
    for (int h = 0; h < 8; ++h) { a[h] = lo[h]; a[h + 8] = hh[h]; }
#pragma unroll
    for (int t = 0; t < 4; ++t) {
      const v16h b = *(const v16h*)(re16 + (size_t)(j0 + t * 16 + mrow) * 64 + kb + hi * 16);
      acc[t] = __builtin_amdgcn_wmma_f32_16x16x32_f16(false, a, false, b,
                                                      (short)0, acc[t], false, false);
    }
  }
#pragma unroll
  for (int t = 0; t < 4; ++t)
#pragma unroll
    for (int r = 0; r < 8; ++r) {
      const int orow = i0 + r + hi * 8;
      const int ocol = j0 + t * 16 + mrow;
      const float v = acc[t][r] * (FINV * FINV);
      out[(size_t)orow * NN + ocol] = __builtin_amdgcn_rcpf(1.0f + __expf(-v));
    }
}

// ---------------------------------------------------------------------------
// Workspace layout (bytes), ~17.9 MB total, all offsets 256B aligned.
// ---------------------------------------------------------------------------
#define OFF_DINV 0u
#define OFF_AW   32768u
#define OFF_BT   (32768u + 8388608u)
#define OFF_H    (OFF_BT + 2097152u)
#define OFF_RE   (OFF_H + 2097152u)
#define OFF_RE16 (OFF_RE + 2097152u)
#define OFF_XD   (OFF_RE16 + 1048576u)

extern "C" void kernel_launch(void* const* d_in, const int* in_sizes, int n_in,
                              void* d_out, int out_size, void* d_ws, size_t ws_size,
                              hipStream_t stream) {
  const float* x   = (const float*)d_in[0];
  const int*   ei  = (const int*)  d_in[1];
  const float* W1  = (const float*)d_in[2];
  const float* b1  = (const float*)d_in[3];
  const float* W2  = (const float*)d_in[4];
  const float* b2  = (const float*)d_in[5];
  const float* We  = (const float*)d_in[6];
  const float* be  = (const float*)d_in[7];
  const float* Wd1 = (const float*)d_in[8];
  const float* bd1 = (const float*)d_in[9];
  const float* Wd2 = (const float*)d_in[10];
  const float* bd2 = (const float*)d_in[11];
  (void)in_sizes; (void)n_in; (void)out_size; (void)ws_size;

  char* ws = (char*)d_ws;
  float*    dinv = (float*)   (ws + OFF_DINV);
  unsigned* aw   = (unsigned*)(ws + OFF_AW);
  _Float16* Bt   = (_Float16*)(ws + OFF_BT);
  float*    hbuf = (float*)   (ws + OFF_H);
  float*    re   = (float*)   (ws + OFF_RE);
  _Float16* re16 = (_Float16*)(ws + OFF_RE16);
  float*    xd   = (float*)   (ws + OFF_XD);

  float* out   = (float*)d_out;
  float* recon = out;                                       // [N,N]
  float* xout  = out + (size_t)NN * NN;                     // [N,128]
  float* z     = out + (size_t)NN * NN + (size_t)NN * 128;  // [N,128]

  // 1) adjacency bit-pack + dinv  (single 256MB scan; 8MB bits live in L2)
  pack_deg_kernel<<<NN, 256, 0, stream>>>(ei, aw, dinv);

  // 2) h = lrelu(nadj (x W1) + b1)          [N,64]
  xw_bt_kernel<128, 64><<<(NN * 64) / 256, 256, 0, stream>>>(x, W1, dinv, Bt);
  spmm_kernel<64><<<NN / 16, 256, 0, stream>>>(aw, Bt, dinv, b1, hbuf, nullptr);

  // 3) z = lrelu(nadj (h W2) + b2)          [N,128]  -> output slot
  xw_bt_kernel<64, 128><<<(NN * 128) / 256, 256, 0, stream>>>(hbuf, W2, dinv, Bt);
  spmm_kernel<128><<<NN / 16, 256, 0, stream>>>(aw, Bt, dinv, b2, z, nullptr);

  // 4) re = lrelu(nadj (z We) + be)         [N,64]  (+ scaled f16 copy)
  xw_bt_kernel<128, 64><<<(NN * 64) / 256, 256, 0, stream>>>(z, We, dinv, Bt);
  spmm_kernel<64><<<NN / 16, 256, 0, stream>>>(aw, Bt, dinv, be, re, re16);

  // 5) recon_edge = sigmoid(re re^T)        [N,N]
  recon_kernel<<<dim3(NN / 256, NN / 16), 128, 0, stream>>>(re16, recon);

  // 6) xd = lrelu(nadj (z Wd1) + bd1)       [N,64]
  xw_bt_kernel<128, 64><<<(NN * 64) / 256, 256, 0, stream>>>(z, Wd1, dinv, Bt);
  spmm_kernel<64><<<NN / 16, 256, 0, stream>>>(aw, Bt, dinv, bd1, xd, nullptr);

  // 7) x_out = lrelu(nadj (xd Wd2) + bd2)   [N,128]
  xw_bt_kernel<64, 128><<<(NN * 128) / 256, 256, 0, stream>>>(xd, Wd2, dinv, Bt);
  spmm_kernel<128><<<NN / 16, 256, 0, stream>>>(aw, Bt, dinv, bd2, xout, nullptr);
}
